// guideEncoder_85409719648565
// MI455X (gfx1250) — matrix operand order repaired and checked by reading, not compiled, not run on hardware
//
#include <hip/hip_runtime.h>

#define N_NODES 50000
#define N_EDGES 800000
#define NFEAT   256
#define NHID    64
#define HEADS   4
#define HF      (HEADS * NHID)   /* 256 */
#define NCLASS  40
#define NEG_SLOPE 0.2f
#define TILE_N  64               /* per-block N extent in GEMM */
#define KMAX    256              /* max K among GEMMs */

typedef __attribute__((ext_vector_type(16))) __bf16       bf16x16;
typedef __attribute__((ext_vector_type(8)))  float        f32x8;
typedef __attribute__((ext_vector_type(4)))  unsigned int u32x4;
typedef __attribute__((ext_vector_type(4)))  int          i32x4;

#ifndef __has_builtin
#define __has_builtin(x) 0
#endif
#if __has_builtin(__builtin_amdgcn_global_load_async_to_lds_b128)
#define HAVE_ASYNC_COPY 1
#else
#define HAVE_ASYNC_COPY 0
#endif

#if HAVE_ASYNC_COPY
typedef __attribute__((address_space(1))) i32x4 g1_i32x4;
typedef __attribute__((address_space(3))) i32x4 lds3_i32x4;
__device__ __forceinline__ void async_copy_b128(const void* g, void* lds) {
  // generic->AS1: same 64-bit value; generic LDS ptr low 32 bits = LDS offset (ISA aperture rule)
  g1_i32x4*   gp = (g1_i32x4*)(unsigned long long)(size_t)g;
  lds3_i32x4* lp = (lds3_i32x4*)(unsigned int)(size_t)lds;
  __builtin_amdgcn_global_load_async_to_lds_b128(gp, lp, 0, 0);
}
__device__ __forceinline__ void async_wait0() {
#if __has_builtin(__builtin_amdgcn_s_wait_asynccnt)
  __builtin_amdgcn_s_wait_asynccnt(0);
#else
  asm volatile("s_wait_asynccnt 0x0" ::: "memory");
#endif
}
#endif

__device__ __forceinline__ unsigned short f2bf(float f) {
  unsigned u = __float_as_uint(f);
  u += 0x7FFFu + ((u >> 16) & 1u);            // round-to-nearest-even
  return (unsigned short)(u >> 16);
}

// ---------------- conversion kernels ----------------
__global__ void k_convert_bf16(const float* __restrict__ in,
                               unsigned short* __restrict__ out, int n) {
  int i = blockIdx.x * blockDim.x + threadIdx.x;
  if (i < n) out[i] = f2bf(in[i]);
}

// W [K,NC] row-major (f32) -> WT [NC,K] row-major (bf16)
__global__ void k_convert_transpose_bf16(const float* __restrict__ W,
                                         unsigned short* __restrict__ WT,
                                         int K, int NC) {
  int i = blockIdx.x * blockDim.x + threadIdx.x;
  if (i >= K * NC) return;
  int n = i / K, k = i % K;
  WT[i] = f2bf(W[(size_t)k * NC + n]);
}

__global__ void k_fill_u32(unsigned int* __restrict__ p, unsigned int v, int n) {
  int i = blockIdx.x * blockDim.x + threadIdx.x;
  if (i < n) p[i] = v;
}

// ---------------- WMMA GEMM: C[M,NC] = A[M,K] * B[K,NC], B given as BT[NC,K] ----------------
// block = 256 threads (8 waves). Block computes a 128 x 64 macro-tile:
//   - B panel (64 cols x K) staged once into LDS (async copy when available)
//   - each wave computes a 16 x 64 strip: 4 accumulators, A fragment reused 4x
__global__ __launch_bounds__(256) void k_wmma_gemm_bf16(
    const unsigned short* __restrict__ A,   // [M,K] bf16 row-major
    const unsigned short* __restrict__ BT,  // [NC,K] bf16 row-major (= B transposed)
    float* __restrict__ C,                  // [M,NC] f32
    int M, int K, int NC) {
  __shared__ unsigned short sB[TILE_N * KMAX];   // 32 KB max

  const int tid  = threadIdx.x;
  const int lane = tid & 31;
  const int wv   = tid >> 5;
  const int N0   = blockIdx.y * TILE_N;

  // ---- stage B panel: BT rows [N0, N0+64) x K -> LDS, 16B chunks ----
  {
    const int chunks = (TILE_N * K) >> 3;          // # of 8-half (16B) chunks
    const unsigned short* gbase = BT + (size_t)N0 * K;
    for (int c = tid; c < chunks; c += 256) {
#if HAVE_ASYNC_COPY
      async_copy_b128(gbase + (size_t)c * 8, &sB[c * 8]);
#else
      *(u32x4*)&sB[c * 8] = *(const u32x4*)(gbase + (size_t)c * 8);
#endif
    }
#if HAVE_ASYNC_COPY
    async_wait0();
#endif
  }
  __syncthreads();

  const int rowTile = blockIdx.x * 8 + wv;         // 16-row tile index
  if (rowTile >= (M >> 4)) return;
  const int M0   = rowTile << 4;
  const int m    = lane & 15;        // A row / B column / C column within tile
  const int half = lane >> 4;        // K sub-phase select

  const unsigned short* arow = A + (size_t)(M0 + m) * K + half * 8;

  f32x8 cc[4] = {{}, {}, {}, {}};
  for (int k0 = 0; k0 < K; k0 += 32) {
    u32x4 a0 = *(const u32x4*)(arow + k0);
    u32x4 a1 = *(const u32x4*)(arow + k0 + 16);
    bf16x16 av = __builtin_bit_cast(bf16x16,
        __builtin_shufflevector(a0, a1, 0, 1, 2, 3, 4, 5, 6, 7));
    #pragma unroll
    for (int nt = 0; nt < 4; ++nt) {
      const unsigned short* bp = &sB[(nt * 16 + m) * K + half * 8 + k0];
      u32x4 b0 = *(const u32x4*)bp;        // ds_load_b128
      u32x4 b1 = *(const u32x4*)(bp + 16); // ds_load_b128
      bf16x16 bv = __builtin_bit_cast(bf16x16,
          __builtin_shufflevector(b0, b1, 0, 1, 2, 3, 4, 5, 6, 7));
      cc[nt] = __builtin_amdgcn_wmma_f32_16x16x32_bf16(false, av, false, bv,
                                                       (short)0, cc[nt], false, false);
    }
  }
  // C: VGPR r -> row M0 + half*8 + r, col N0 + nt*16 + (lane&15)
  float* cbase = C + (size_t)(M0 + half * 8) * NC + N0 + m;
  #pragma unroll
  for (int nt = 0; nt < 4; ++nt)
    #pragma unroll
    for (int r = 0; r < 8; ++r)
      cbase[(size_t)r * NC + nt * 16] = cc[nt][r];
}

// ---------------- attention scores: alpha_s/alpha_d [N,H] ----------------
__global__ void k_att_scores(const float* __restrict__ h,      // [N, H*F]
                             const float* __restrict__ a_src,  // [H,F]
                             const float* __restrict__ a_dst,  // [H,F]
                             float* __restrict__ as, float* __restrict__ ad) {
  int i = blockIdx.x * blockDim.x + threadIdx.x;
  if (i >= N_NODES * HEADS) return;
  int node = i / HEADS, hd = i % HEADS;
  const float* hp  = h + (size_t)node * HF + hd * NHID;
  const float* sp  = a_src + hd * NHID;
  const float* dp  = a_dst + hd * NHID;
  float s1 = 0.f, s2 = 0.f;
  #pragma unroll 4
  for (int f = 0; f < NHID; ++f) { float v = hp[f]; s1 += v * sp[f]; s2 += v * dp[f]; }
  as[i] = s1; ad[i] = s2;
}

// ---------------- edge pass 1: leaky-relu score + segment max (sign-split atomic trick) ----------------
__global__ void k_edge_score_max(const int* __restrict__ src, const int* __restrict__ dst,
                                 const float* __restrict__ as, const float* __restrict__ ad,
                                 float* __restrict__ ebuf, float* __restrict__ mb) {
  int i = blockIdx.x * blockDim.x + threadIdx.x;
  if (i >= N_EDGES * HEADS) return;
  int e = i / HEADS, hd = i % HEADS;
  int s = src[e], d = dst[e];
  float v = as[s * HEADS + hd] + ad[d * HEADS + hd];
  v = v > 0.f ? v : NEG_SLOPE * v;
  ebuf[i] = v;
  int off = d * HEADS + hd;
  if (v >= 0.f) atomicMax((int*)mb + off, __float_as_int(v));
  else          atomicMin((unsigned int*)mb + off, (unsigned int)__float_as_int(v));
}

// ---------------- edge pass 2: exp(e - m[dst]) and segment sum ----------------
__global__ void k_edge_softmax(const int* __restrict__ dst,
                               const float* __restrict__ mb,
                               float* __restrict__ ebuf, float* __restrict__ dn) {
  int i = blockIdx.x * blockDim.x + threadIdx.x;
  if (i >= N_EDGES * HEADS) return;
  int e = i / HEADS, hd = i % HEADS;
  int off = dst[e] * HEADS + hd;
  float ex = __expf(ebuf[i] - mb[off]);
  ebuf[i] = ex;
  atomicAdd(&dn[off], ex);
}

// ---------------- edge pass 3: scatter alpha * h[src] into acc[dst] ----------------
__global__ __launch_bounds__(HF) void k_edge_scatter(
    const int* __restrict__ src, const int* __restrict__ dst,
    const float* __restrict__ ew, const float* __restrict__ ebuf,
    const float* __restrict__ dn, const float* __restrict__ h,
    float* __restrict__ acc) {
  int e = blockIdx.x;
  int t = threadIdx.x;          // t = hd*64 + f
  int hd = t >> 6;
  int s = src[e], d = dst[e];
  float alpha = ebuf[(size_t)e * HEADS + hd]
              / (dn[d * HEADS + hd] + 1e-16f) * ew[e];
  atomicAdd(&acc[(size_t)d * HF + t], alpha * h[(size_t)s * HF + t]);
}

// ---------------- head mean + bias: [N,H,F] -> [N,F] ----------------
__global__ void k_head_mean(const float* __restrict__ acc,
                            const float* __restrict__ bias,
                            float* __restrict__ out) {
  int i = blockIdx.x * blockDim.x + threadIdx.x;
  if (i >= N_NODES * NHID) return;
  int node = i >> 6, f = i & 63;
  const float* p = acc + (size_t)node * HF;
  out[i] = 0.25f * (p[f] + p[NHID + f] + p[2 * NHID + f] + p[3 * NHID + f]) + bias[f];
}

// ---------------- LayerNorm -> ReLU -> fc ----------------
__global__ __launch_bounds__(NHID) void k_ln_relu_fc(
    const float* __restrict__ hf, const float* __restrict__ g,
    const float* __restrict__ b,  const float* __restrict__ fcW,  // [NHID,NCLASS]
    const float* __restrict__ fcb, float* __restrict__ out) {
  __shared__ float red[NHID];
  __shared__ float sh[NHID];
  int n = blockIdx.x, f = threadIdx.x;
  float v = hf[(size_t)n * NHID + f];
  red[f] = v; __syncthreads();
  for (int s = NHID / 2; s > 0; s >>= 1) { if (f < s) red[f] += red[f + s]; __syncthreads(); }
  float mu = red[0] * (1.f / NHID); __syncthreads();
  float d = v - mu;
  red[f] = d * d; __syncthreads();
  for (int s = NHID / 2; s > 0; s >>= 1) { if (f < s) red[f] += red[f + s]; __syncthreads(); }
  float var = red[0] * (1.f / NHID);
  float nv = d * rsqrtf(var + 1e-5f) * g[f] + b[f];
  sh[f] = nv > 0.f ? nv : 0.f;
  __syncthreads();
  if (f < NCLASS) {
    float a = fcb[f];
    #pragma unroll 8
    for (int j = 0; j < NHID; ++j) a += sh[j] * fcW[j * NCLASS + f];
    out[(size_t)n * NCLASS + f] = a;
  }
}

// =======================================================================
extern "C" void kernel_launch(void* const* d_in, const int* in_sizes, int n_in,
                              void* d_out, int out_size, void* d_ws, size_t ws_size,
                              hipStream_t stream) {
  (void)in_sizes; (void)n_in; (void)out_size; (void)ws_size;

  const float* x      = (const float*)d_in[0];
  const int*   ei     = (const int*)d_in[1];
  const float* ew     = (const float*)d_in[2];
  const float* W1     = (const float*)d_in[3];
  const float* a_src1 = (const float*)d_in[4];
  const float* a_dst1 = (const float*)d_in[5];
  const float* b1     = (const float*)d_in[6];
  const float* W2     = (const float*)d_in[7];
  const float* a_src2 = (const float*)d_in[8];
  const float* a_dst2 = (const float*)d_in[9];
  const float* b2     = (const float*)d_in[10];
  const float* ln_g   = (const float*)d_in[11];
  const float* ln_b   = (const float*)d_in[12];
  const float* fc_W   = (const float*)d_in[13];
  const float* fc_b   = (const float*)d_in[14];
  float* out = (float*)d_out;

  const int* src = ei;
  const int* dst = ei + N_EDGES;

  // ---- carve workspace (256B aligned blocks), ~176 MB total ----
  char* wp = (char*)d_ws;
  auto carve = [&](size_t bytes) -> void* {
    void* p = (void*)wp;
    wp += (bytes + 255) & ~(size_t)255;
    return p;
  };
  unsigned short* xb   = (unsigned short*)carve((size_t)N_NODES * NFEAT * 2);
  unsigned short* w1t  = (unsigned short*)carve((size_t)NFEAT * HF * 2);    // [HF, NFEAT]
  unsigned short* w2t  = (unsigned short*)carve((size_t)NHID * HF * 2);     // [HF, NHID]
  float* h    = (float*)carve((size_t)N_NODES * HF * 4);                    // h1, then h2
  float* as   = (float*)carve((size_t)N_NODES * HEADS * 4);
  float* ad   = (float*)carve((size_t)N_NODES * HEADS * 4);
  float* mb   = (float*)carve((size_t)N_NODES * HEADS * 4);
  float* dn   = (float*)carve((size_t)N_NODES * HEADS * 4);
  float* eb   = (float*)carve((size_t)N_EDGES * HEADS * 4);
  float* acc  = (float*)carve((size_t)N_NODES * HF * 4);
  float* hm1  = (float*)carve((size_t)N_NODES * NHID * 4);
  unsigned short* hm1b = (unsigned short*)carve((size_t)N_NODES * NHID * 2);
  float* hm2  = (float*)carve((size_t)N_NODES * NHID * 4);

  auto grid1d = [](int n, int blk) { return dim3((unsigned)((n + blk - 1) / blk)); };
  const int B = 256;
  const dim3 gemmGrid((unsigned)(((N_NODES >> 4) + 7) / 8), HF / TILE_N);

  // ---- precision conversion (bf16 WMMA path) ----
  k_convert_bf16<<<grid1d(N_NODES * NFEAT, B), B, 0, stream>>>(x, xb, N_NODES * NFEAT);
  k_convert_transpose_bf16<<<grid1d(NFEAT * HF, B), B, 0, stream>>>(W1, w1t, NFEAT, HF);
  k_convert_transpose_bf16<<<grid1d(NHID * HF, B), B, 0, stream>>>(W2, w2t, NHID, HF);

  // ---- layer 1 ----
  {
    k_wmma_gemm_bf16<<<gemmGrid, B, 0, stream>>>(xb, w1t, h, N_NODES, NFEAT, HF);

    k_att_scores<<<grid1d(N_NODES * HEADS, B), B, 0, stream>>>(h, a_src1, a_dst1, as, ad);

    k_fill_u32<<<grid1d(N_NODES * HEADS, B), B, 0, stream>>>((unsigned*)mb, 0xFF800000u, N_NODES * HEADS); // -inf
    k_fill_u32<<<grid1d(N_NODES * HEADS, B), B, 0, stream>>>((unsigned*)dn, 0u, N_NODES * HEADS);
    k_fill_u32<<<grid1d(N_NODES * HF, B), B, 0, stream>>>((unsigned*)acc, 0u, N_NODES * HF);

    k_edge_score_max<<<grid1d(N_EDGES * HEADS, B), B, 0, stream>>>(src, dst, as, ad, eb, mb);
    k_edge_softmax<<<grid1d(N_EDGES * HEADS, B), B, 0, stream>>>(dst, mb, eb, dn);
    k_edge_scatter<<<dim3(N_EDGES), dim3(HF), 0, stream>>>(src, dst, ew, eb, dn, h, acc);
    k_head_mean<<<grid1d(N_NODES * NHID, B), B, 0, stream>>>(acc, b1, hm1);
  }

  // ---- layer 2 ----
  {
    k_convert_bf16<<<grid1d(N_NODES * NHID, B), B, 0, stream>>>(hm1, hm1b, N_NODES * NHID);
    k_wmma_gemm_bf16<<<gemmGrid, B, 0, stream>>>(hm1b, w2t, h, N_NODES, NHID, HF);

    k_att_scores<<<grid1d(N_NODES * HEADS, B), B, 0, stream>>>(h, a_src2, a_dst2, as, ad);

    k_fill_u32<<<grid1d(N_NODES * HEADS, B), B, 0, stream>>>((unsigned*)mb, 0xFF800000u, N_NODES * HEADS);
    k_fill_u32<<<grid1d(N_NODES * HEADS, B), B, 0, stream>>>((unsigned*)dn, 0u, N_NODES * HEADS);
    k_fill_u32<<<grid1d(N_NODES * HF, B), B, 0, stream>>>((unsigned*)acc, 0u, N_NODES * HF);

    k_edge_score_max<<<grid1d(N_EDGES * HEADS, B), B, 0, stream>>>(src, dst, as, ad, eb, mb);
    k_edge_softmax<<<grid1d(N_EDGES * HEADS, B), B, 0, stream>>>(dst, mb, eb, dn);
    k_edge_scatter<<<dim3(N_EDGES), dim3(HF), 0, stream>>>(src, dst, ew, eb, dn, h, acc);
    k_head_mean<<<grid1d(N_NODES * NHID, B), B, 0, stream>>>(acc, b2, hm2);
  }

  // ---- LayerNorm -> ReLU -> Linear ----
  k_ln_relu_fc<<<dim3(N_NODES), dim3(NHID), 0, stream>>>(hm2, ln_g, ln_b, fc_W, fc_b, out);
}
